// PRISM_80788334838235
// MI455X (gfx1250) — matrix operand order repaired and checked by reading, not compiled, not run on hardware
//
#include <hip/hip_runtime.h>
#include <hip/hip_bf16.h>
#include <math.h>

// ---------------------------------------------------------------------------
// Problem constants (match the reference)
// ---------------------------------------------------------------------------
#define B_    64
#define S_    512
#define D_    512
#define H_    8
#define DK_   64
#define P_    16
#define FF_   2048
#define L_    3
#define BS_   (B_ * S_)       // 32768 rows
#define FREQ_ 257
#define CUT_  85              // FREQ // 3
#define EPS_  1e-5f

typedef unsigned short us;                                    // bf16 storage
typedef __attribute__((ext_vector_type(16))) __bf16 v16bf;    // WMMA A/B frag
typedef __attribute__((ext_vector_type(8)))  float  v8f;      // WMMA C/D frag

// ---------------------------------------------------------------------------
// bf16 helpers (round-to-nearest-even), pure bit ops (no __bf16 arithmetic)
// ---------------------------------------------------------------------------
__device__ __forceinline__ us f2bf(float f) {
  union { float f; unsigned u; } v; v.f = f;
  unsigned r = (v.u + 0x7FFFu + ((v.u >> 16) & 1u)) >> 16;
  return (us)r;
}
__device__ __forceinline__ float bf2f(us h) {
  union { unsigned u; float f; } v; v.u = ((unsigned)h) << 16;
  return v.f;
}

// ---------------------------------------------------------------------------
// gfx1250 async global->LDS copy (ISA §10 / §15.18.3, tracked by ASYNCcnt).
// Per-lane: LDS[vdst] <= MEM[vaddr], 16 bytes. LDS byte address is the low
// 32 bits of the generic pointer (LDS aperture maps addr[31:0]).
// ---------------------------------------------------------------------------
__device__ __forceinline__ void async_b128(const void* gsrc, void* lds_dst) {
  unsigned l = (unsigned)(unsigned long long)lds_dst;
  asm volatile("global_load_async_to_lds_b128 %0, %1, off"
               :: "v"(l), "v"(gsrc) : "memory");
}
__device__ __forceinline__ void wait_async0() {
  asm volatile("s_wait_asynccnt 0x0" ::: "memory");
}

// ---------------------------------------------------------------------------
// WMMA fragment loaders. 16-bit K order used here (contiguous, not the ISA table order):
//  A 16x32 (MxK): lane<16 -> row=lane, K 0..15 in elems 0..15
//                 lane>=16 -> row=lane-16, K 16..31
//  B 32x16 (KxN): lane n = lane&15; lanes<16 hold K 0..15, lanes>=16 K 16..31
//                 (contiguous in K when B is stored K-fastest, i.e. [n][k])
//  C/D 16x16:     VGPR v: lanes0-15 (M=v, N=lane); lanes16-31 (M=v+8, N=lane-16)
// ---------------------------------------------------------------------------
__device__ __forceinline__ v16bf frag_a16(const us* base, int stride, int lane) {
  const us* p = base + (size_t)(lane & 15) * stride + ((lane >> 4) << 4);
  uint4 lo = *(const uint4*)p;
  uint4 hi = *(const uint4*)(p + 8);
  v16bf f;
  __builtin_memcpy(&f, &lo, 16);
  __builtin_memcpy((char*)&f + 16, &hi, 16);
  return f;
}
__device__ __forceinline__ v16bf frag_b16(const us* base, int stride, int lane) {
  const us* p = base + (size_t)(lane & 15) * stride + ((lane >> 4) << 4);
  uint4 lo = *(const uint4*)p;
  uint4 hi = *(const uint4*)(p + 8);
  v16bf f;
  __builtin_memcpy(&f, &lo, 16);
  __builtin_memcpy((char*)&f + 16, &hi, 16);
  return f;
}
// Gather variant for strided B (element (k,n) at base + k*sk + n*sn)
__device__ __forceinline__ v16bf frag_b_gen(const us* base, size_t sk, size_t sn, int lane) {
  int n = lane & 15;
  int k0 = (lane >> 4) << 4;
  v16bf f;
#pragma unroll
  for (int e = 0; e < 16; ++e) {
    us u = base[(size_t)(k0 + e) * sk + (size_t)n * sn];
    __bf16 bb; __builtin_memcpy(&bb, &u, 2);
    f[e] = bb;
  }
  return f;
}
__device__ __forceinline__ v8f wmma_bf(v16bf a, v16bf b, v8f c) {
  // emits v_wmma_f32_16x16x32_bf16
  return __builtin_amdgcn_wmma_f32_16x16x32_bf16(false, a, false, b, (short)0, c,
                                                 false, false);
}

template <int ACT> __device__ __forceinline__ float epilogue_act(float x) {
  if (ACT == 1) return 0.5f * x * (1.0f + erff(x * 0.70710678118f));   // exact GELU
  if (ACT == 2) return 1.0f / (1.0f + __expf(-x));                    // sigmoid
  return x;
}

// ---------------------------------------------------------------------------
// GEMM: C[M,N] = act(A[M,K]_bf16 * B[K,N]_bf16 + bias[N]);  out f32 or bf16.
// Tile 128x128x32, 256 thr (8 waves), wave = 32x64 => 2x4 wmma tiles.
// Double-buffered LDS; A staged by async global->LDS (ASYNCcnt), B staged
// one tile ahead through registers (transposed to [n][k] in LDS).
// ---------------------------------------------------------------------------
#define BM 128
#define BN 128
#define BKK 32

template <bool OUTBF, int ACT>
__global__ __launch_bounds__(256) void gemm_bf16_k(
    const us* __restrict__ A, const us* __restrict__ Bw,
    const float* __restrict__ bias, void* __restrict__ C,
    int M, int N, int K) {
  __shared__ __align__(16) us As[2][BM * BKK];
  __shared__ __align__(16) us Bs[2][BN * BKK];   // transposed: [n][k]
  const int tid = threadIdx.x;
  const int lane = tid & 31;
  const int wave = tid >> 5;
  const int bm = blockIdx.y * BM;
  const int bn = blockIdx.x * BN;
  const int wm = (wave >> 1) * 32;   // wave row offset in tile
  const int wn = (wave & 1) * 64;    // wave col offset in tile

  v8f acc[2][4];
  v8f zz = {};
#pragma unroll
  for (int i = 0; i < 2; ++i)
#pragma unroll
    for (int j = 0; j < 4; ++j) acc[i][j] = zz;

  us breg[16];
  auto stageA = [&](int k0, int bsel) {   // async global->LDS, 2 x b128/thread
#pragma unroll
    for (int i = 0; i < 2; ++i) {
      int e = tid + i * 256;
      int r = e >> 2, g = e & 3;          // 4 groups of 8 bf16 per row
      async_b128(&A[(size_t)(bm + r) * K + k0 + g * 8],
                 &As[bsel][r * BKK + g * 8]);
    }
  };
  auto loadB = [&](int k0) {              // 16 scalar bf16/thread into regs
#pragma unroll
    for (int j = 0; j < 16; ++j) {
      int e = tid + j * 256;
      int kk = e >> 7, n = e & 127;
      breg[j] = Bw[(size_t)(k0 + kk) * N + bn + n];
    }
  };
  auto storeB = [&](int bsel) {           // transpose into LDS [n][k]
#pragma unroll
    for (int j = 0; j < 16; ++j) {
      int e = tid + j * 256;
      int kk = e >> 7, n = e & 127;
      Bs[bsel][n * BKK + kk] = breg[j];
    }
  };

  const int T = K / BKK;
  stageA(0, 0);
  loadB(0);
  storeB(0);
  wait_async0();
  __syncthreads();

  for (int t = 0; t < T; ++t) {
    const int cur = t & 1, nxt = cur ^ 1;
    if (t + 1 < T) {                      // prefetch next tile while computing
      stageA((t + 1) * BKK, nxt);
      loadB((t + 1) * BKK);
    }
    if (t + 2 < T) {                      // gfx1250 global_prefetch_b8, t+2
      __builtin_prefetch(&A[(size_t)(bm + (tid >> 1)) * K + (t + 2) * BKK], 0, 1);
      __builtin_prefetch(&Bw[(size_t)((t + 2) * BKK + (tid & 31)) * N + bn], 0, 1);
    }

    v16bf af0 = frag_a16(&As[cur][(wm + 0) * BKK], BKK, lane);
    v16bf af1 = frag_a16(&As[cur][(wm + 16) * BKK], BKK, lane);
#pragma unroll
    for (int ni = 0; ni < 4; ++ni) {
      v16bf bfr = frag_b16(&Bs[cur][(wn + ni * 16) * BKK], BKK, lane);
      acc[0][ni] = wmma_bf(af0, bfr, acc[0][ni]);
      acc[1][ni] = wmma_bf(af1, bfr, acc[1][ni]);
    }
    if (t + 1 < T) storeB(nxt);
    wait_async0();
    __syncthreads();
  }

  const int lm = lane & 15, lh = lane >> 4;
#pragma unroll
  for (int mi = 0; mi < 2; ++mi)
#pragma unroll
    for (int ni = 0; ni < 4; ++ni)
#pragma unroll
      for (int v = 0; v < 8; ++v) {
        int row = bm + wm + mi * 16 + lh * 8 + v;
        int col = bn + wn + ni * 16 + lm;
        float x = acc[mi][ni][v] + bias[col];
        x = epilogue_act<ACT>(x);
        if (OUTBF) ((us*)C)[(size_t)row * N + col] = f2bf(x);
        else       ((float*)C)[(size_t)row * N + col] = x;
      }
}

// ---------------------------------------------------------------------------
// Flash attention (non-causal), one block per (64 q rows, head, batch).
// 4 waves; wave owns 16 q rows. QK^T and PV on WMMA, online softmax in regs.
// Q/K tiles staged by async global->LDS.
// ---------------------------------------------------------------------------
__global__ __launch_bounds__(128) void flash_attn_k(
    const us* __restrict__ Qb, const us* __restrict__ Kb,
    const us* __restrict__ Vb, us* __restrict__ Ob) {
  __shared__ __align__(16) us Qs[64 * 64];
  __shared__ __align__(16) us Ks[64 * 64];        // [kpos][dk]
  __shared__ __align__(16) us Vst[64 * 64];       // transposed: [dk][kpos]
  __shared__ __align__(16) us Ps[4][16 * 64];     // per-wave P scratch
  const int tid = threadIdx.x, lane = tid & 31, w = tid >> 5;
  const int q0 = blockIdx.x * 64;
  const int h = blockIdx.y;
  const int b = blockIdx.z;
  const int lm = lane & 15, lh = lane >> 4;
  const float scale = 0.125f;   // 1/sqrt(64)

  const size_t baseQ = ((size_t)b * S_ + q0) * D_ + h * DK_;
  for (int e = tid; e < 64 * 8; e += 128) {
    int r = e >> 3, g = e & 7;
    async_b128(&Qb[baseQ + (size_t)r * D_ + g * 8], &Qs[r * 64 + g * 8]);
  }
  wait_async0();
  __syncthreads();
  v16bf aQ0 = frag_a16(&Qs[(w * 16) * 64 + 0], 64, lane);
  v16bf aQ1 = frag_a16(&Qs[(w * 16) * 64 + 32], 64, lane);

  v8f o[4];
  v8f zz = {};
  float mrow[8], lrow[8];
#pragma unroll
  for (int i = 0; i < 4; ++i) o[i] = zz;
#pragma unroll
  for (int v = 0; v < 8; ++v) { mrow[v] = -1e30f; lrow[v] = 0.f; }

  for (int kb = 0; kb < S_; kb += 64) {
    const size_t baseK = ((size_t)b * S_ + kb) * D_ + h * DK_;
    for (int e = tid; e < 64 * 8; e += 128) {
      int r = e >> 3, g = e & 7;
      async_b128(&Kb[baseK + (size_t)r * D_ + g * 8], &Ks[r * 64 + g * 8]);
    }
    for (int e = tid; e < 64 * 64; e += 128) {     // V transposed for B frags
      int r = e >> 6, n = e & 63;
      Vst[n * 64 + r] = Vb[baseK + (size_t)r * D_ + n];
    }
    wait_async0();
    __syncthreads();

    v8f sc[4];
#pragma unroll
    for (int nt = 0; nt < 4; ++nt) {
      v8f c = zz;
      c = wmma_bf(aQ0, frag_b16(&Ks[(nt * 16) * 64 + 0], 64, lane), c);
      c = wmma_bf(aQ1, frag_b16(&Ks[(nt * 16) * 64 + 32], 64, lane), c);
      sc[nt] = c;
    }

    us* psw = &Ps[w][0];
#pragma unroll
    for (int v = 0; v < 8; ++v) {
      float s0 = sc[0][v] * scale, s1 = sc[1][v] * scale;
      float s2 = sc[2][v] * scale, s3 = sc[3][v] * scale;
      float mx = fmaxf(fmaxf(s0, s1), fmaxf(s2, s3));
#pragma unroll
      for (int off = 1; off < 16; off <<= 1) mx = fmaxf(mx, __shfl_xor(mx, off, 32));
      float mnew = fmaxf(mrow[v], mx);
      float alpha = __expf(mrow[v] - mnew);
      float p0 = __expf(s0 - mnew), p1 = __expf(s1 - mnew);
      float p2 = __expf(s2 - mnew), p3 = __expf(s3 - mnew);
      float ps = p0 + p1 + p2 + p3;
#pragma unroll
      for (int off = 1; off < 16; off <<= 1) ps += __shfl_xor(ps, off, 32);
      lrow[v] = lrow[v] * alpha + ps;
      mrow[v] = mnew;
      o[0][v] *= alpha; o[1][v] *= alpha; o[2][v] *= alpha; o[3][v] *= alpha;
      int rloc = lh * 8 + v;
      psw[rloc * 64 + 0 + lm]  = f2bf(p0);
      psw[rloc * 64 + 16 + lm] = f2bf(p1);
      psw[rloc * 64 + 32 + lm] = f2bf(p2);
      psw[rloc * 64 + 48 + lm] = f2bf(p3);
    }
    __builtin_amdgcn_wave_barrier();     // P in LDS, per-wave RAW ordering

    v16bf aP0 = frag_a16(psw + 0, 64, lane);
    v16bf aP1 = frag_a16(psw + 32, 64, lane);
#pragma unroll
    for (int nd = 0; nd < 4; ++nd) {
      o[nd] = wmma_bf(aP0, frag_b16(&Vst[(nd * 16) * 64 + 0], 64, lane), o[nd]);
      o[nd] = wmma_bf(aP1, frag_b16(&Vst[(nd * 16) * 64 + 32], 64, lane), o[nd]);
    }
    __syncthreads();
  }

#pragma unroll
  for (int nd = 0; nd < 4; ++nd)
#pragma unroll
    for (int v = 0; v < 8; ++v) {
      int q = q0 + w * 16 + lh * 8 + v;
      int dcol = nd * 16 + lm;
      Ob[((size_t)b * S_ + q) * D_ + h * DK_ + dcol] = f2bf(o[nd][v] / lrow[v]);
    }
}

// ---------------------------------------------------------------------------
// Primitive decomposition: per (h,b) block, 256 thr.
//   sp = Qp Kp^T * scale (WMMA) ; gscores += rowsums ; Ap = softmax(sp)
//   pf(head slice) = Ap Vp (WMMA, K split across waves, LDS reduce)
// ---------------------------------------------------------------------------
__global__ __launch_bounds__(256) void prim_attn_k(
    const us* __restrict__ pqb,   // [P,D] bf16
    const us* __restrict__ Kpb,   // [B,S,D] bf16
    const us* __restrict__ Vpb,   // [B,S,D] bf16
    float* __restrict__ pf,       // [B,P,D]
    float* __restrict__ gsc) {    // [B,P] (atomic accum)
  __shared__ __align__(16) float sp[P_ * S_];   // 32 KB, reused as reducer
  __shared__ __align__(16) us ap[P_ * S_];      // 16 KB
  const int tid = threadIdx.x, lane = tid & 31, w = tid >> 5;
  const int h = blockIdx.x, b = blockIdx.y;
  const int lm = lane & 15, lh = lane >> 4;
  const float scale = 0.125f;
  v8f zz = {};

  {
    v16bf a0 = frag_a16(&pqb[h * DK_ + 0], D_, lane);
    v16bf a1 = frag_a16(&pqb[h * DK_ + 32], D_, lane);
    const int s0 = w * 64;
#pragma unroll
    for (int nt = 0; nt < 4; ++nt) {
      const us* kb = &Kpb[((size_t)b * S_ + s0 + nt * 16) * D_ + h * DK_];
      v8f c = zz;
      c = wmma_bf(a0, frag_b16(kb + 0, D_, lane), c);
      c = wmma_bf(a1, frag_b16(kb + 32, D_, lane), c);
#pragma unroll
      for (int v = 0; v < 8; ++v) {
        int p = lh * 8 + v;
        sp[p * S_ + s0 + nt * 16 + lm] = c[v] * scale;
      }
    }
  }
  __syncthreads();

  // row softmax + gscore partial (wave w handles rows w and w+8)
  for (int rr = 0; rr < 2; ++rr) {
    int p = w + rr * 8;
    float vals[16];
    float mx = -1e30f, tot = 0.f;
#pragma unroll
    for (int i = 0; i < 16; ++i) {
      float v = sp[p * S_ + lane + i * 32];
      vals[i] = v; mx = fmaxf(mx, v); tot += v;
    }
#pragma unroll
    for (int off = 1; off < 32; off <<= 1) {
      mx = fmaxf(mx, __shfl_xor(mx, off, 32));
      tot += __shfl_xor(tot, off, 32);
    }
    float sm = 0.f;
#pragma unroll
    for (int i = 0; i < 16; ++i) { vals[i] = __expf(vals[i] - mx); sm += vals[i]; }
#pragma unroll
    for (int off = 1; off < 32; off <<= 1) sm += __shfl_xor(sm, off, 32);
    float inv = 1.0f / sm;
#pragma unroll
    for (int i = 0; i < 16; ++i)
      ap[p * S_ + lane + i * 32] = f2bf(vals[i] * inv);
    if (lane == 0) atomicAdd(&gsc[b * P_ + p], tot * (1.0f / (float)(H_ * S_)));
  }
  __syncthreads();

  // pf = Ap (16xS) * Vp (S x 64) ; wave w handles K slices {2w, 2w+1}
  v8f acc[4];
#pragma unroll
  for (int i = 0; i < 4; ++i) acc[i] = zz;
  for (int t = 0; t < 2; ++t) {
    int kk = w * 2 + t;
    v16bf aA = frag_a16(&ap[kk * 32], S_, lane);
#pragma unroll
    for (int nt = 0; nt < 4; ++nt) {
      v16bf bV = frag_b_gen(
          &Vpb[((size_t)b * S_ + kk * 32) * D_ + h * DK_ + nt * 16], D_, 1, lane);
      acc[nt] = wmma_bf(aA, bV, acc[nt]);
    }
  }
  __syncthreads();
  float* red = sp;   // reuse as [8][16][64]
#pragma unroll
  for (int nt = 0; nt < 4; ++nt)
#pragma unroll
    for (int v = 0; v < 8; ++v) {
      int p = lh * 8 + v, n = nt * 16 + lm;
      red[(w * 16 + p) * 64 + n] = acc[nt][v];
    }
  __syncthreads();
  for (int e = tid; e < P_ * DK_; e += 256) {
    int p = e >> 6, n = e & 63;
    float ssum = 0.f;
#pragma unroll
    for (int ww = 0; ww < 8; ++ww) ssum += red[(ww * 16 + p) * 64 + n];
    pf[((size_t)b * P_ + p) * D_ + h * DK_ + n] = ssum;
  }
}

// ---------------------------------------------------------------------------
// Spectral filter: forward FFT(512), weight by fw (Hermitian-extended),
// inverse(full) and inverse(low-mask); xh = full - low. 1 wave per channel.
// ---------------------------------------------------------------------------
__device__ __forceinline__ void fft512_stages(float* wr_, float* wi_, int lane,
                                              float sgn) {
  for (int len = 2; len <= 512; len <<= 1) {
    int half = len >> 1;
    float ang0 = sgn * 6.28318530718f / (float)len;
    for (int t = lane; t < 256; t += 32) {
      int grp = t / half, pos = t - grp * half;
      int i0 = grp * len + pos, i1 = i0 + half;
      float ss, cc;
      __sincosf(ang0 * (float)pos, &ss, &cc);
      float ur = wr_[i0], ui = wi_[i0];
      float vr = wr_[i1], vi = wi_[i1];
      float tr = vr * cc - vi * ss, ti = vr * ss + vi * cc;
      wr_[i0] = ur + tr; wi_[i0] = ui + ti;
      wr_[i1] = ur - tr; wi_[i1] = ui - ti;
    }
    __syncthreads();
  }
}

__global__ __launch_bounds__(128) void fft_filter_k(
    const float* __restrict__ x, const float* __restrict__ fw,
    float* __restrict__ xl, float* __restrict__ xh) {
  __shared__ float wr[4][512], wi[4][512];   // work
  __shared__ float sr[4][512], si[4][512];   // weighted spectrum
  const int lane = threadIdx.x & 31, w = threadIdx.x >> 5;
  const int d = blockIdx.x * 4 + w;
  const int b = blockIdx.y;
  float keep[16];

  // forward: bit-reversed load of real signal
  for (int i = 0; i < 16; ++i) {
    int s = lane + i * 32;
    int r = __brev((unsigned)s) >> 23;
    wr[w][r] = x[((size_t)b * S_ + s) * D_ + d];
    wi[w][r] = 0.f;
  }
  __syncthreads();
  fft512_stages(&wr[w][0], &wi[w][0], lane, -1.0f);

  for (int i = 0; i < 16; ++i) {
    int k = lane + i * 32;
    int kk = (k <= 256) ? k : (512 - k);
    float wgt = fw[kk];
    sr[w][k] = wr[w][k] * wgt;
    si[w][k] = wi[w][k] * wgt;
  }
  __syncthreads();

  // inverse of full weighted spectrum
  for (int i = 0; i < 16; ++i) {
    int k = lane + i * 32;
    int r = __brev((unsigned)k) >> 23;
    wr[w][r] = sr[w][k]; wi[w][r] = si[w][k];
  }
  __syncthreads();
  fft512_stages(&wr[w][0], &wi[w][0], lane, 1.0f);
  for (int i = 0; i < 16; ++i) keep[i] = wr[w][lane + i * 32] * (1.f / 512.f);
  __syncthreads();

  // inverse of low-masked spectrum
  for (int i = 0; i < 16; ++i) {
    int k = lane + i * 32;
    int r = __brev((unsigned)k) >> 23;
    bool low = (k < CUT_) || (k > 512 - CUT_);
    wr[w][r] = low ? sr[w][k] : 0.f;
    wi[w][r] = low ? si[w][k] : 0.f;
  }
  __syncthreads();
  fft512_stages(&wr[w][0], &wi[w][0], lane, 1.0f);
  for (int i = 0; i < 16; ++i) {
    int s = lane + i * 32;
    float lo = wr[w][s] * (1.f / 512.f);
    xl[((size_t)b * S_ + s) * D_ + d] = lo;
    xh[((size_t)b * S_ + s) * D_ + d] = keep[i] - lo;
  }
}

// ---------------------------------------------------------------------------
// Fused residual + LayerNorm, dual f32/bf16 output. One block per row.
// MODE 0: y = x + r0[row]          (tensor residual)
// MODE 1: y = x + r0[b]            (row-broadcast residual)
// MODE 2: y = x + g*r1 + (1-g)*r2  (gate combine; g = r0)
// ---------------------------------------------------------------------------
template <int MODE>
__global__ __launch_bounds__(256) void ln_fuse_k(
    const float* __restrict__ xin, const float* __restrict__ r0,
    const float* __restrict__ r1, const float* __restrict__ r2,
    const float* __restrict__ gamma, const float* __restrict__ beta,
    float* __restrict__ outf, us* __restrict__ outb) {
  __shared__ float sred[16];
  __shared__ float smv[2];
  const int row = blockIdx.x;
  const int tid = threadIdx.x;
  const size_t base = (size_t)row * D_;
  float y[2];
#pragma unroll
  for (int j = 0; j < 2; ++j) {
    int i = tid + j * 256;
    float v = xin[base + i];
    if (MODE == 0) v += r0[base + i];
    else if (MODE == 1) v += r0[(size_t)(row / S_) * D_ + i];
    else {
      float g = r0[base + i];
      v += g * r1[base + i] + (1.f - g) * r2[base + i];
    }
    y[j] = v;
  }
  float s1 = y[0] + y[1], s2 = y[0] * y[0] + y[1] * y[1];
#pragma unroll
  for (int off = 1; off < 32; off <<= 1) {
    s1 += __shfl_xor(s1, off, 32);
    s2 += __shfl_xor(s2, off, 32);
  }
  int lane = tid & 31, w = tid >> 5;
  if (lane == 0) { sred[w] = s1; sred[8 + w] = s2; }
  __syncthreads();
  if (tid == 0) {
    float a = 0.f, c = 0.f;
    for (int i = 0; i < 8; ++i) { a += sred[i]; c += sred[8 + i]; }
    float mean = a * (1.0f / D_);
    float var = c * (1.0f / D_) - mean * mean;
    smv[0] = mean; smv[1] = rsqrtf(var + EPS_);
  }
  __syncthreads();
  float mean = smv[0], inv = smv[1];
#pragma unroll
  for (int j = 0; j < 2; ++j) {
    int i = tid + j * 256;
    float o = (y[j] - mean) * inv * gamma[i] + beta[i];
    outf[base + i] = o;
    outb[base + i] = f2bf(o);
  }
}

// ---------------------------------------------------------------------------
// Small helper kernels
// ---------------------------------------------------------------------------
__global__ void cvt_bf16_k(const float* __restrict__ in, us* __restrict__ out,
                           int n, int rowLen, int outStride, int colOff) {
  int i = blockIdx.x * 256 + threadIdx.x;
  if (i >= n) return;
  int r = i / rowLen, c = i - r * rowLen;
  out[(size_t)r * outStride + colOff + c] = f2bf(in[i]);
}

__global__ void zero_k(float* p, int n) {
  int i = blockIdx.x * 256 + threadIdx.x;
  if (i < n) p[i] = 0.f;
}

__global__ __launch_bounds__(32) void prim_softmax_k(
    const float* __restrict__ gsc, float* __restrict__ pw,
    float* __restrict__ pwOut) {
  int b = blockIdx.x, lane = threadIdx.x;
  float v = (lane < P_) ? gsc[b * P_ + lane] : -1e30f;
  float m = v;
#pragma unroll
  for (int off = 1; off < 32; off <<= 1) m = fmaxf(m, __shfl_xor(m, off, 32));
  float e = (lane < P_) ? __expf(v - m) : 0.f;
  float s = e;
#pragma unroll
  for (int off = 1; off < 32; off <<= 1) s += __shfl_xor(s, off, 32);
  if (lane < P_) {
    float r = e / s;
    pw[b * P_ + lane] = r;
    pwOut[b * P_ + lane] = r;
  }
}

__global__ __launch_bounds__(256) void agg_k(const float* __restrict__ pw,
                                             const float* __restrict__ pf,
                                             float* __restrict__ agg) {
  int b = blockIdx.x;
  for (int dIdx = threadIdx.x; dIdx < D_; dIdx += 256) {
    float s = 0.f;
#pragma unroll
    for (int p = 0; p < P_; ++p)
      s += pw[b * P_ + p] * pf[((size_t)b * P_ + p) * D_ + dIdx];
    agg[(size_t)b * D_ + dIdx] = s;
  }
}

// out[b,n] = agg[b,:] @ W[:,n] + bias[n]    (tiny GEMM, 16.7 MFLOP)
__global__ __launch_bounds__(256) void rowgemm_k(
    const float* __restrict__ A, const float* __restrict__ W,
    const float* __restrict__ bias, float* __restrict__ out) {
  int idx = blockIdx.x * 256 + threadIdx.x;
  int b = idx >> 9, n = idx & 511;
  float s = bias[n];
  for (int k = 0; k < D_; ++k) s += A[b * D_ + k] * W[(size_t)k * D_ + n];
  out[idx] = s;
}

__global__ __launch_bounds__(256) void divloss_k(const float* __restrict__ pf,
                                                 float* __restrict__ divOut) {
  __shared__ float pfs[P_ * D_];   // 32 KB
  __shared__ float nrm[P_];
  __shared__ float part[8];
  int b = blockIdx.x, tid = threadIdx.x;
  for (int e = tid; e < P_ * D_; e += 256)
    pfs[e] = pf[(size_t)b * P_ * D_ + e];
  __syncthreads();
  if (tid < P_) {
    float s = 0.f;
    for (int k = 0; k < D_; ++k) { float v = pfs[tid * D_ + k]; s += v * v; }
    nrm[tid] = rsqrtf(s);
  }
  __syncthreads();
  int p = tid >> 4, q = tid & 15;
  float contrib = 0.f;
  if (p != q) {
    float dt = 0.f;
    for (int k = 0; k < D_; ++k) dt += pfs[p * D_ + k] * pfs[q * D_ + k];
    contrib = fabsf(dt * nrm[p] * nrm[q]);
  }
#pragma unroll
  for (int off = 1; off < 32; off <<= 1) contrib += __shfl_xor(contrib, off, 32);
  if ((tid & 31) == 0) part[tid >> 5] = contrib;
  __syncthreads();
  if (tid == 0) {
    float s = 0.f;
    for (int i = 0; i < 8; ++i) s += part[i];
    atomicAdd(divOut, s * (1.0f / (float)(B_ * P_ * P_)));
  }
}

// ---------------------------------------------------------------------------
// Orchestration. Workspace requirement ~0.84 GB.
// ---------------------------------------------------------------------------
extern "C" void kernel_launch(void* const* d_in, const int* in_sizes, int n_in,
                              void* d_out, int out_size, void* d_ws,
                              size_t ws_size, hipStream_t stream) {
  (void)in_sizes; (void)n_in; (void)out_size; (void)ws_size;
  const float* x_in = (const float*)d_in[0];
  const float* Wq = (const float*)d_in[1];
  const float* Wk = (const float*)d_in[2];
  const float* Wv = (const float*)d_in[3];
  const float* Wo = (const float*)d_in[4];
  const float* bq = (const float*)d_in[5];
  const float* bk = (const float*)d_in[6];
  const float* bv = (const float*)d_in[7];
  const float* bo = (const float*)d_in[8];
  const float* g1 = (const float*)d_in[9];
  const float* be1 = (const float*)d_in[10];
  const float* pq = (const float*)d_in[11];
  const float* Wpk = (const float*)d_in[12];
  const float* Wpv = (const float*)d_in[13];
  const float* Wpo = (const float*)d_in[14];
  const float* bpk = (const float*)d_in[15];
  const float* bpv = (const float*)d_in[16];
  const float* bpo = (const float*)d_in[17];
  const float* gp = (const float*)d_in[18];
  const float* bp = (const float*)d_in[19];
  const float* fw = (const float*)d_in[20];
  const float* Wl = (const float*)d_in[21];
  const float* bl = (const float*)d_in[22];
  const float* Wh = (const float*)d_in[23];
  const float* bh = (const float*)d_in[24];
  const float* Wg = (const float*)d_in[25];
  const float* bg = (const float*)d_in[26];
  const float* gs = (const float*)d_in[27];
  const float* bs = (const float*)d_in[28];
  const float* W1 = (const float*)d_in[29];
  const float* b1 = (const float*)d_in[30];
  const float* W2 = (const float*)d_in[31];
  const float* b2 = (const float*)d_in[32];
  const float* g2 = (const float*)d_in[33];
  const float* be2 = (const float*)d_in[34];

  char* ws = (char*)d_ws;
  size_t off = 0;
  auto alloc = [&](size_t bytes) -> void* {
    void* p = ws + off;
    off += (bytes + 255) & ~(size_t)255;
    return p;
  };
  const size_t TBYT = (size_t)BS_ * D_ * 4;   // 64 MiB per f32 tensor
  const size_t HBYT = (size_t)BS_ * D_ * 2;   // 32 MiB per bf16 tensor
  float* F_X  = (float*)alloc(TBYT);
  float* F_T0 = (float*)alloc(TBYT);
  float* F_T1 = (float*)alloc(TBYT);
  float* F_T2 = (float*)alloc(TBYT);
  float* F_T3 = (float*)alloc(TBYT);
  float* F_T4 = (float*)alloc(TBYT);
  float* F_PF = (float*)alloc((size_t)B_ * P_ * D_ * 4);
  float* F_GSC = (float*)alloc((size_t)B_ * P_ * 4);
  float* F_PW  = (float*)alloc((size_t)B_ * P_ * 4);
  float* F_AGG = (float*)alloc((size_t)B_ * D_ * 4);
  float* F_OUTP = (float*)alloc((size_t)B_ * D_ * 4);
  us* H_XB = (us*)alloc(HBYT);
  us* H_QB = (us*)alloc(HBYT);
  us* H_KB = (us*)alloc(HBYT);
  us* H_VB = (us*)alloc(HBYT);
  us* H_OB = (us*)alloc(HBYT);
  us* H_LB = (us*)alloc(HBYT);
  us* H_HB = (us*)alloc(HBYT);
  us* H_CAT = (us*)alloc((size_t)BS_ * 2 * D_ * 2);   // [BS, 2D] bf16
  us* H_FF  = (us*)alloc((size_t)BS_ * FF_ * 2);      // [BS, FF] bf16
  us* H_WB  = (us*)alloc((size_t)FF_ * D_ * 2);       // weight staging
  us* H_PQB = (us*)alloc((size_t)P_ * D_ * 2);

  float* out_x = (float*)d_out;
  float* divPtr = out_x + (size_t)BS_ * D_;
  float* out_pw = divPtr + 1;

  auto cvt = [&](const float* in, us* outp, int n, int rowLen, int stride,
                 int colOff) {
    cvt_bf16_k<<<dim3((n + 255) / 256), dim3(256), 0, stream>>>(
        in, outp, n, rowLen, stride, colOff);
  };

  const dim3 gN512(D_ / BN, BS_ / BM);        // (4, 256)
  const dim3 gNFF(FF_ / BN, BS_ / BM);        // (16, 256)
  const dim3 blk256(256), blk128(128);

  // init
  zero_k<<<1, 256, 0, stream>>>(divPtr, 1);
  hipMemcpyAsync(F_X, x_in, TBYT, hipMemcpyDeviceToDevice, stream);
  cvt(x_in, H_XB, BS_ * D_, D_, D_, 0);

  const int DDsz = D_ * D_;
  for (int i = 0; i < L_; ++i) {
    const float* Wq_i = Wq + (size_t)i * DDsz;  const float* bq_i = bq + i * D_;
    const float* Wk_i = Wk + (size_t)i * DDsz;  const float* bk_i = bk + i * D_;
    const float* Wv_i = Wv + (size_t)i * DDsz;  const float* bv_i = bv + i * D_;
    const float* Wo_i = Wo + (size_t)i * DDsz;  const float* bo_i = bo + i * D_;

    // ---- multi-head self attention ----
    cvt(Wq_i, H_WB, DDsz, DDsz, DDsz, 0);
    gemm_bf16_k<true, 0><<<gN512, blk256, 0, stream>>>(H_XB, H_WB, bq_i, H_QB,
                                                       BS_, D_, D_);
    cvt(Wk_i, H_WB, DDsz, DDsz, DDsz, 0);
    gemm_bf16_k<true, 0><<<gN512, blk256, 0, stream>>>(H_XB, H_WB, bk_i, H_KB,
                                                       BS_, D_, D_);
    cvt(Wv_i, H_WB, DDsz, DDsz, DDsz, 0);
    gemm_bf16_k<true, 0><<<gN512, blk256, 0, stream>>>(H_XB, H_WB, bv_i, H_VB,
                                                       BS_, D_, D_);
    flash_attn_k<<<dim3(S_ / 64, H_, B_), blk128, 0, stream>>>(H_QB, H_KB, H_VB,
                                                               H_OB);
    cvt(Wo_i, H_WB, DDsz, DDsz, DDsz, 0);
    gemm_bf16_k<false, 0><<<gN512, blk256, 0, stream>>>(H_OB, H_WB, bo_i, F_T0,
                                                        BS_, D_, D_);
    ln_fuse_k<0><<<dim3(BS_), blk256, 0, stream>>>(
        F_X, F_T0, nullptr, nullptr, g1 + i * D_, be1 + i * D_, F_X, H_XB);

    // ---- primitive dictionary decomposition ----
    cvt(Wpk + (size_t)i * DDsz, H_WB, DDsz, DDsz, DDsz, 0);
    gemm_bf16_k<true, 0><<<gN512, blk256, 0, stream>>>(
        H_XB, H_WB, bpk + i * D_, H_QB /*Kp*/, BS_, D_, D_);
    cvt(Wpv + (size_t)i * DDsz, H_WB, DDsz, DDsz, DDsz, 0);
    gemm_bf16_k<true, 0><<<gN512, blk256, 0, stream>>>(
        H_XB, H_WB, bpv + i * D_, H_KB /*Vp*/, BS_, D_, D_);
    cvt(pq + (size_t)i * P_ * D_, H_PQB, P_ * D_, P_ * D_, P_ * D_, 0);
    zero_k<<<dim3((B_ * P_ + 255) / 256), blk256, 0, stream>>>(F_GSC, B_ * P_);
    prim_attn_k<<<dim3(H_, B_), blk256, 0, stream>>>(H_PQB, H_QB, H_KB, F_PF,
                                                     F_GSC);
    prim_softmax_k<<<dim3(B_), dim3(32), 0, stream>>>(F_GSC, F_PW, out_pw);
    agg_k<<<dim3(B_), blk256, 0, stream>>>(F_PW, F_PF, F_AGG);
    rowgemm_k<<<dim3((B_ * D_) / 256), blk256, 0, stream>>>(
        F_AGG, Wpo + (size_t)i * DDsz, bpo + i * D_, F_OUTP);
    ln_fuse_k<1><<<dim3(BS_), blk256, 0, stream>>>(
        F_X, F_OUTP, nullptr, nullptr, gp + i * D_, bp + i * D_, F_X, H_XB);
    divloss_k<<<dim3(B_), blk256, 0, stream>>>(F_PF, divPtr);

    // ---- adaptive spectral refinement ----
    fft_filter_k<<<dim3(D_ / 4, B_), blk128, 0, stream>>>(
        F_X, fw + (size_t)i * FREQ_, F_T0 /*xl*/, F_T1 /*xh*/);
    cvt(F_T0, H_LB, BS_ * D_, D_, D_, 0);
    cvt(F_T1, H_HB, BS_ * D_, D_, D_, 0);
    cvt(Wl + (size_t)i * DDsz, H_WB, DDsz, DDsz, DDsz, 0);
    gemm_bf16_k<false, 0><<<gN512, blk256, 0, stream>>>(
        H_LB, H_WB, bl + i * D_, F_T2, BS_, D_, D_);
    cvt(Wh + (size_t)i * DDsz, H_WB, DDsz, DDsz, DDsz, 0);
    gemm_bf16_k<false, 0><<<gN512, blk256, 0, stream>>>(
        H_HB, H_WB, bh + i * D_, F_T3, BS_, D_, D_);
    cvt(F_T2, H_CAT, BS_ * D_, D_, 2 * D_, 0);
    cvt(F_T3, H_CAT, BS_ * D_, D_, 2 * D_, D_);
    cvt(Wg + (size_t)i * 2 * DDsz, H_WB, 2 * DDsz, 2 * DDsz, 2 * DDsz, 0);
    gemm_bf16_k<false, 2><<<gN512, blk256, 0, stream>>>(
        H_CAT, H_WB, bg + i * D_, F_T4 /*gate*/, BS_, D_, 2 * D_);
    ln_fuse_k<2><<<dim3(BS_), blk256, 0, stream>>>(
        F_X, F_T4, F_T2, F_T3, gs + i * D_, bs + i * D_, F_X, H_XB);

    // ---- position-wise FFN ----
    cvt(W1 + (size_t)i * D_ * FF_, H_WB, D_ * FF_, D_ * FF_, D_ * FF_, 0);
    gemm_bf16_k<true, 1><<<gNFF, blk256, 0, stream>>>(
        H_XB, H_WB, b1 + i * FF_, H_FF, BS_, FF_, D_);
    cvt(W2 + (size_t)i * FF_ * D_, H_WB, FF_ * D_, FF_ * D_, FF_ * D_, 0);
    gemm_bf16_k<false, 0><<<gN512, blk256, 0, stream>>>(
        H_FF, H_WB, b2 + i * D_, F_T0, BS_, D_, FF_);
    ln_fuse_k<0><<<dim3(BS_), blk256, 0, stream>>>(
        F_X, F_T0, nullptr, nullptr, g2 + i * D_, be2 + i * D_, F_X, H_XB);
  }

  hipMemcpyAsync(out_x, F_X, TBYT, hipMemcpyDeviceToDevice, stream);
}
